// UoPAttention_85701777425143
// MI455X (gfx1250) — matrix operand order repaired and checked by reading, not compiled, not run on hardware
//
#include <hip/hip_runtime.h>
#include <hip/hip_bf16.h>
#include <math.h>

#define B_  64
#define S_  4096
#define D_  256
#define P_  256

typedef __attribute__((ext_vector_type(16))) __bf16 v16bf;
typedef __attribute__((ext_vector_type(8)))  float  v8f;

union Frag16 { v16bf v; unsigned short u[16]; unsigned d[8]; };

__device__ __forceinline__ unsigned short f2bf(float f) {
    union { float f; unsigned u; } c; c.f = f;
    unsigned u = c.u;
    u += 0x7fffu + ((u >> 16) & 1u);   // round-to-nearest-even
    return (unsigned short)(u >> 16);
}

// Pack two f32 -> packed bf16x2 dword. Prefer HW packed convert on gfx1250.
__device__ __forceinline__ unsigned pk_bf16(float x, float y) {
#if __has_builtin(__builtin_amdgcn_cvt_pk_bf16_f32)
    auto v = __builtin_amdgcn_cvt_pk_bf16_f32(x, y);   // v_cvt_pk_bf16_f32
    return __builtin_bit_cast(unsigned, v);
#else
    return (unsigned)f2bf(x) | ((unsigned)f2bf(y) << 16);
#endif
}

__device__ __forceinline__ float fast_tanh(float x) {
#if __has_builtin(__builtin_amdgcn_tanhf)
    return __builtin_amdgcn_tanhf(x);                  // v_tanh_f32 (TRANS)
#else
    float e = __expf(2.0f * x);
    return 1.0f - 2.0f / (e + 1.0f);
#endif
}

// ---------------------------------------------------------------------------
// a[b,p] = input_1[b,:] . W_attr[p,:] + b_attr[p] + b_text[p]   (tiny GEMM)
// ---------------------------------------------------------------------------
__global__ void attr_proj_kernel(const float* __restrict__ in1,
                                 const float* __restrict__ W_attr,
                                 const float* __restrict__ b_attr,
                                 const float* __restrict__ b_text,
                                 float* __restrict__ a_ws) {
    const int b = blockIdx.x, p = threadIdx.x;
    const float* x = in1 + b * D_;
    const float* w = W_attr + p * D_;
    float acc = b_attr[p] + b_text[p];
#pragma unroll 8
    for (int d = 0; d < D_; ++d) acc += x[d] * w[d];
    a_ws[b * P_ + p] = acc;
}

// ---------------------------------------------------------------------------
// Fused: t = in2 @ W_text^T (bf16 WMMA), h = tanh(a + t), score = h . w_pool
// LDS: W_text bf16 [P][D] (128KB) + a-row (1KB) + w_pool (1KB).
// Block = 256 thr (8 waves); block covers one batch b, 256 S rows.
// Each wave owns 32 rows = TWO 16-row A-tile sets held in registers, and
// issues two WMMAs per B-fragment LDS load (B shared across both tiles),
// halving LDS read traffic per WMMA.
// ---------------------------------------------------------------------------
__global__ void __launch_bounds__(256) score_kernel(
    const float* __restrict__ in2,
    const float* __restrict__ Wtext,
    const float* __restrict__ a_ws,
    const float* __restrict__ w_pool,
    const float* __restrict__ b_pool,
    float* __restrict__ scores) {
    extern __shared__ __align__(16) unsigned char smem[];
    unsigned short* lds_wt = (unsigned short*)smem;           // [P_][D_] bf16
    float*          lds_a  = (float*)(smem + P_ * D_ * 2);    // [P_]
    float*          lds_wp = lds_a + P_;                      // [P_]

    const int tid   = threadIdx.x;
    const int b     = blockIdx.x >> 4;   // 16 chunks of 256 rows per batch
    const int chunk = blockIdx.x & 15;

    // Stage W_text -> LDS bf16 (coalesced float2 loads, packed b32 stores)
    {
        unsigned* dst = (unsigned*)lds_wt;
        const float2* src = (const float2*)Wtext;
#pragma unroll 4
        for (int i = 0; i < 128; ++i) {
            const float2 w2 = src[i * 256 + tid];
            dst[i * 256 + tid] = pk_bf16(w2.x, w2.y);
        }
        lds_a[tid]  = a_ws[b * P_ + tid];
        lds_wp[tid] = w_pool[tid];
    }
    __syncthreads();

    const int wave = tid >> 5;
    const int lane = tid & 31;
    const int lh   = lane & 15;        // lane within half
    const int hi   = lane >> 4;        // which 16-lane half

    const float bp = b_pool[0];

    // This wave's 32 rows: two 16-row tiles at s0 and s0+16.
    const int s0 = chunk * 256 + wave * 32;
    const float* row0 = in2 + ((size_t)(b * S_ + s0 + lh)) * D_;
    const float* row1 = row0 + (size_t)16 * D_;

    // A fragments (16x32 bf16 per k-step), documented lane layout:
    // halfs[0..7]  = K = kb*32 + hi*8 + (0..7)
    // halfs[8..15] = K = kb*32 + 16 + hi*8 + (0..7)
    Frag16 Af0[8], Af1[8];
#pragma unroll
    for (int kb = 0; kb < 8; ++kb) {
        {
            const float4* q = (const float4*)(row0 + kb * 32 + hi * 8);
            const float4 q0 = q[0], q1 = q[1], q2 = q[4], q3 = q[5];
            Af0[kb].d[0] = pk_bf16(q0.x, q0.y);
            Af0[kb].d[1] = pk_bf16(q0.z, q0.w);
            Af0[kb].d[2] = pk_bf16(q1.x, q1.y);
            Af0[kb].d[3] = pk_bf16(q1.z, q1.w);
            Af0[kb].d[4] = pk_bf16(q2.x, q2.y);
            Af0[kb].d[5] = pk_bf16(q2.z, q2.w);
            Af0[kb].d[6] = pk_bf16(q3.x, q3.y);
            Af0[kb].d[7] = pk_bf16(q3.z, q3.w);
        }
        {
            const float4* q = (const float4*)(row1 + kb * 32 + hi * 8);
            const float4 q0 = q[0], q1 = q[1], q2 = q[4], q3 = q[5];
            Af1[kb].d[0] = pk_bf16(q0.x, q0.y);
            Af1[kb].d[1] = pk_bf16(q0.z, q0.w);
            Af1[kb].d[2] = pk_bf16(q1.x, q1.y);
            Af1[kb].d[3] = pk_bf16(q1.z, q1.w);
            Af1[kb].d[4] = pk_bf16(q2.x, q2.y);
            Af1[kb].d[5] = pk_bf16(q2.z, q2.w);
            Af1[kb].d[6] = pk_bf16(q3.x, q3.y);
            Af1[kb].d[7] = pk_bf16(q3.z, q3.w);
        }
    }

    float part0[8] = {0.f, 0.f, 0.f, 0.f, 0.f, 0.f, 0.f, 0.f};
    float part1[8] = {0.f, 0.f, 0.f, 0.f, 0.f, 0.f, 0.f, 0.f};

#pragma unroll 1
    for (int nb = 0; nb < 16; ++nb) {
        v8f acc0 = {};
        v8f acc1 = {};
        const int prow = nb * 16 + lh;  // W_text row p (= B column n)
#pragma unroll
        for (int kb = 0; kb < 8; ++kb) {
            Frag16 Bf;
            // B 32x16: lane holds K-halfs hi*8+(0..7) and 16+hi*8+(0..7) of column prow
            const uint4* bsrc =
                (const uint4*)(lds_wt + prow * 256 + kb * 32 + hi * 8);
            ((uint4*)Bf.u)[0] = bsrc[0];
            ((uint4*)Bf.u)[1] = bsrc[2];
            acc0 = __builtin_amdgcn_wmma_f32_16x16x32_bf16(
                false, Af0[kb].v, false, Bf.v, (short)0, acc0, false, false);
            acc1 = __builtin_amdgcn_wmma_f32_16x16x32_bf16(
                false, Af1[kb].v, false, Bf.v, (short)0, acc1, false, false);
        }
        // Fused epilogue: h = tanh(acc + a[p]); part += h * w_pool[p]
        const int p    = nb * 16 + lh;
        const float av = lds_a[p];
        const float wp = lds_wp[p];
#pragma unroll
        for (int r = 0; r < 8; ++r) {
            part0[r] += fast_tanh(acc0[r] + av) * wp;
            part1[r] += fast_tanh(acc1[r] + av) * wp;
        }
    }

    // C layout: VGPR r -> row s0+r (lanes 0-15) / s0+8+r (lanes 16-31).
    // Reduce the 16 p-columns held across each 16-lane half.
#pragma unroll
    for (int r = 0; r < 8; ++r) {
        float v0 = part0[r];
        v0 += __shfl_xor(v0, 1, 32);
        v0 += __shfl_xor(v0, 2, 32);
        v0 += __shfl_xor(v0, 4, 32);
        v0 += __shfl_xor(v0, 8, 32);
        float v1 = part1[r];
        v1 += __shfl_xor(v1, 1, 32);
        v1 += __shfl_xor(v1, 2, 32);
        v1 += __shfl_xor(v1, 4, 32);
        v1 += __shfl_xor(v1, 8, 32);
        if (lh == 0) {
            scores[(size_t)b * S_ + s0 + hi * 8 + r]      = v0 + bp;
            scores[(size_t)b * S_ + s0 + 16 + hi * 8 + r] = v1 + bp;
        }
    }
}

// ---------------------------------------------------------------------------
// Masked softmax over S per batch, in place on the weights region of d_out.
// ---------------------------------------------------------------------------
__global__ void softmax_kernel(const int* __restrict__ mask,
                               float* __restrict__ wts) {
    const int b = blockIdx.x;
    const int tid = threadIdx.x;
    __shared__ float red[256];
    float*      w = wts  + (size_t)b * S_;
    const int*  m = mask + (size_t)b * S_;

    float lv[16];
    float mx = -1e30f;
#pragma unroll
    for (int i = 0; i < 16; ++i) {
        const int s = tid + i * 256;
        float v = w[s];
        v = (m[s] == 0) ? -1e9f : v;
        lv[i] = v;
        mx = fmaxf(mx, v);
    }
    red[tid] = mx; __syncthreads();
    for (int off = 128; off > 0; off >>= 1) {
        if (tid < off) red[tid] = fmaxf(red[tid], red[tid + off]);
        __syncthreads();
    }
    mx = red[0]; __syncthreads();

    float sum = 0.f;
#pragma unroll
    for (int i = 0; i < 16; ++i) {
        const float e = __expf(lv[i] - mx);
        lv[i] = e;
        sum += e;
    }
    red[tid] = sum; __syncthreads();
    for (int off = 128; off > 0; off >>= 1) {
        if (tid < off) red[tid] += red[tid + off];
        __syncthreads();
    }
    const float inv = 1.0f / red[0];
#pragma unroll
    for (int i = 0; i < 16; ++i)
        w[tid + i * 256] = lv[i] * inv;
}

// ---------------------------------------------------------------------------
// pooled[b,d] = sum_s weights[b,s] * in2[b,s,d] — two-stage (deterministic).
// ---------------------------------------------------------------------------
__global__ void pool_partial_kernel(const float* __restrict__ in2,
                                    const float* __restrict__ wts,
                                    float* __restrict__ partial) {
    const int b = blockIdx.y, chunk = blockIdx.x, d = threadIdx.x;
    const float* base = in2 + ((size_t)b * S_ + chunk * 256) * D_;
    const float* w    = wts + (size_t)b * S_ + chunk * 256;
    float acc = 0.f;
#pragma unroll 8
    for (int s = 0; s < 256; ++s)
        acc += w[s] * base[(size_t)s * D_ + d];
    partial[((size_t)b * 16 + chunk) * D_ + d] = acc;
}

__global__ void pool_reduce_kernel(const float* __restrict__ partial,
                                   float* __restrict__ pooled) {
    const int b = blockIdx.x, d = threadIdx.x;
    float acc = 0.f;
#pragma unroll
    for (int c = 0; c < 16; ++c)
        acc += partial[((size_t)b * 16 + c) * D_ + d];
    pooled[b * D_ + d] = acc;
}

// ---------------------------------------------------------------------------
extern "C" void kernel_launch(void* const* d_in, const int* in_sizes, int n_in,
                              void* d_out, int out_size, void* d_ws, size_t ws_size,
                              hipStream_t stream) {
    const float* in1   = (const float*)d_in[0];
    const float* in2   = (const float*)d_in[1];
    const int*   mask  = (const int*)  d_in[2];
    const float* Wattr = (const float*)d_in[3];
    const float* battr = (const float*)d_in[4];
    const float* Wtext = (const float*)d_in[5];
    const float* btext = (const float*)d_in[6];
    const float* wpool = (const float*)d_in[7];
    const float* bpool = (const float*)d_in[8];

    float* out    = (float*)d_out;
    float* pooled = out;                // [B, D]
    float* wts    = out + B_ * D_;      // [B, S]: scores then weights in place

    float* a_ws    = (float*)d_ws;      // [B, P]
    float* partial = a_ws + B_ * P_;    // [B, 16, D]

    const int lds_bytes = P_ * D_ * (int)sizeof(unsigned short)  // W_text bf16
                        + 2 * P_ * (int)sizeof(float);           // a row + w_pool
    (void)hipFuncSetAttribute((const void*)score_kernel,
                              hipFuncAttributeMaxDynamicSharedMemorySize,
                              lds_bytes);

    attr_proj_kernel<<<dim3(B_), dim3(P_), 0, stream>>>(in1, Wattr, battr, btext, a_ws);
    score_kernel<<<dim3(B_ * 16), dim3(256), lds_bytes, stream>>>(
        in2, Wtext, a_ws, wpool, bpool, wts);
    softmax_kernel<<<dim3(B_), dim3(256), 0, stream>>>(mask, wts);
    pool_partial_kernel<<<dim3(16, B_), dim3(D_), 0, stream>>>(in2, wts, partial);
    pool_reduce_kernel<<<dim3(B_), dim3(D_), 0, stream>>>(partial, pooled);
}
